// Discriminator_42107859370041
// MI455X (gfx1250) — matrix operand order repaired and checked by reading, not compiled, not run on hardware
//
#include <hip/hip_runtime.h>
#include <hip/hip_bf16.h>
#include <cmath>

// GRU: B=64, T=1024, S=1024. bf16 WMMA; persistent-kernel recurrence.
#define B_ 64
#define T_ 1024
#define S_ 1024
#define NWG 64        // persistent workgroups; each owns 16 hidden units
#define UNITS 16
#define KTILES (S_ / 32)
#define XG_NT 4       // N-tiles per wave in the x-gates GEMM

typedef __bf16 bf16;
typedef bf16  v16bf __attribute__((ext_vector_type(16)));
typedef bf16  v8bf  __attribute__((ext_vector_type(8)));
typedef float v8f   __attribute__((ext_vector_type(8)));
typedef float v4f   __attribute__((ext_vector_type(4)));

// Workspace layout (bytes):
//   [0,256)                 : grid-barrier counter
//   [256, +3*B*S*2)         : 3 ring buffers of h (bf16)          (384 KB)
//   [WS_W_OFF, +12.58MB)    : packed bf16 weights (W_ih & W_hh)
//   [WS_XG_OFF, +384MB)     : precomputed x-gates (bf16), only if ws is big
#define WS_CTR_OFF   0
#define WS_H_OFF     256
#define WS_H_BYTES   (3 * B_ * S_ * 2)
#define WS_W_OFF     (WS_H_OFF + WS_H_BYTES)
#define WPACK_ELEMS  (2ull * 3 * NWG * KTILES * 512)   // 6,291,456 bf16
#define WPACK_BYTES  (WPACK_ELEMS * 2)
#define WS_XG_OFF    (WS_W_OFF + WPACK_BYTES)
#define XG_ELEMS     ((size_t)T_ * NWG * 3 * B_ * UNITS)       // 201,326,592
#define XG_BYTES     (XG_ELEMS * 2)                            // 384 MB
#define WS_NEED_BIG  (WS_XG_OFF + XG_BYTES)

// LDS staging sizes (dynamic shared memory; CDNA5: up to 320 KB per WG)
#define WSLICE_ELEMS (3 * KTILES * 512)                 // one proj, one wg: 49152
#define GATE_F32     (6 * B_ * UNITS)                   // 6144 floats (24 KB)

__device__ __forceinline__ float sigmoidf_(float x) {
    return 1.0f / (1.0f + __expf(-x));
}

// B-operand tile: 32 lanes x 16 bf16, contiguous (two b128 loads per lane)
__device__ __forceinline__ v16bf load_b_tile(const bf16* bp) {
    v8bf b0 = *(const v8bf*)bp;
    v8bf b1 = *(const v8bf*)(bp + 8);
    v16bf bv;
#pragma unroll
    for (int e = 0; e < 8; ++e) { bv[e] = b0[e]; bv[8 + e] = b1[e]; }
    return bv;
}

// A-operand (16x32 bf16 tile) from an fp32 row-major source
__device__ __forceinline__ v16bf load_a_f32(const float* __restrict__ row, int k0) {
    v4f f0 = *(const v4f*)(row + k0);
    v4f f1 = *(const v4f*)(row + k0 + 4);
    v4f f2 = *(const v4f*)(row + k0 + 16);
    v4f f3 = *(const v4f*)(row + k0 + 20);
    v16bf a;
#pragma unroll
    for (int e = 0; e < 4; ++e) {
        a[e]      = (bf16)f0[e];
        a[4 + e]  = (bf16)f1[e];
        a[8 + e]  = (bf16)f2[e];
        a[12 + e] = (bf16)f3[e];
    }
    return a;
}

// A-operand (16x32 bf16 tile) from a bf16 row-major source
__device__ __forceinline__ v16bf load_a_bf16(const bf16* __restrict__ row, int k0) {
    v8bf l0 = *(const v8bf*)(row + k0);
    v8bf l1 = *(const v8bf*)(row + k0 + 16);
    v16bf a;
#pragma unroll
    for (int e = 0; e < 8; ++e) { a[e] = l0[e]; a[8 + e] = l1[e]; }
    return a;
}

#define WMMA_BF16(A, Bv, C) \
    __builtin_amdgcn_wmma_f32_16x16x32_bf16(false, (A), false, (Bv), (short)0, (C), false, false)

// ---------------------------------------------------------------------------
// Pack W_ih / W_hh (fp32 [3S x S]) into WMMA-ready bf16 B-tiles.
//   tile id = ((proj*3+gate)*NWG + wg); per ktile: 32 lanes x 16 bf16
//   n=lane&15, h=lane>>4: k = kt*32 + 8h + i (i<8) ; k = kt*32 + 16 + 8h + (i-8) (i>=8)
// ---------------------------------------------------------------------------
__global__ void pack_weights(const float* __restrict__ Wih,
                             const float* __restrict__ Whh,
                             bf16* __restrict__ wpack) {
    size_t idx = (size_t)blockIdx.x * 256 + threadIdx.x;
    if (idx >= WPACK_ELEMS) return;
    int i    = (int)(idx & 15);
    int lane = (int)((idx >> 4) & 31);
    int kt   = (int)((idx >> 9) & 31);
    int wg   = (int)((idx >> 14) & 63);
    int pg   = (int)(idx >> 20);            // 0..5 = proj*3+gate
    int proj = pg / 3, gate = pg % 3;
    int n = lane & 15;
    int k = kt * 32 + ((i < 8) ? (8 * (lane >> 4) + i) : (16 + 8 * (lane >> 4) + (i - 8)));
    int grow = gate * S_ + wg * UNITS + n;
    const float* W = proj ? Whh : Wih;
    wpack[idx] = (bf16)W[(size_t)grow * S_ + k];
}

__global__ void init_state(bf16* __restrict__ hbuf, unsigned* __restrict__ ctr) {
    size_t i = (size_t)blockIdx.x * 256 + threadIdx.x;
    if (i < (size_t)3 * B_ * S_) hbuf[i] = (bf16)0.0f;
    if (i == 0) *ctr = 0u;
}

// ---------------------------------------------------------------------------
// Phase 1 (big-ws path): x_gates = batch @ W_ih^T + b_ih, bf16 output.
// One wave per (16x16 M-tile) x (4 N-tiles): one A load/convert feeds 4 WMMAs.
// A tile software-pipelined (a_next) to break the WAR hazard between the
// fp32->bf16 converts and the in-flight WMMA reading the A registers;
// 4 B tiles hoisted to locals so their 8 loads can issue as one clause.
// Output layout: xg[t][wg][gate][b][n] (bf16), bias folded in.
// ---------------------------------------------------------------------------
__launch_bounds__(256, 2)
__global__ void xg_gemm(const float* __restrict__ batch,
                        const float* __restrict__ b_ih,
                        const bf16* __restrict__ wpack,
                        bf16* __restrict__ xg) {
    const int wave = threadIdx.x >> 5;
    const int lane = threadIdx.x & 31;
    const int half = lane >> 4;
    const size_t task = (size_t)blockIdx.x * 8 + wave;     // < 4096 * 48
    const int ng    = (int)(task % (192 / XG_NT));         // group of 4 ntiles
    const int mtile = (int)(task / (192 / XG_NT));
    const int nt0   = ng * XG_NT;
    const int gate  = nt0 / NWG;          // same gate for all 4 (64 % 4 == 0)
    const int wg0   = nt0 % NWG;
    const int b     = mtile / (T_ / 16);
    const int tbase = (mtile % (T_ / 16)) * 16;
    const size_t wtile = (size_t)KTILES * 512;

    const float* arow = batch + ((size_t)b * T_ + tbase + (lane & 15)) * S_;
    const bf16*  wt0  = wpack + ((size_t)gate * NWG + wg0) * wtile + lane * 16;

    v8f acc0 = {}, acc1 = {}, acc2 = {}, acc3 = {};
    v16bf a = load_a_f32(arow, half * 8);
    for (int kt = 0; kt < KTILES; ++kt) {
        const bf16* wk = wt0 + (size_t)kt * 512;
        v16bf bv0 = load_b_tile(wk + 0 * wtile);
        v16bf bv1 = load_b_tile(wk + 1 * wtile);
        v16bf bv2 = load_b_tile(wk + 2 * wtile);
        v16bf bv3 = load_b_tile(wk + 3 * wtile);
        v16bf a_next = a;
        if (kt + 1 < KTILES) a_next = load_a_f32(arow, (kt + 1) * 32 + half * 8);
        acc0 = WMMA_BF16(a, bv0, acc0);
        acc1 = WMMA_BF16(a, bv1, acc1);
        acc2 = WMMA_BF16(a, bv2, acc2);
        acc3 = WMMA_BF16(a, bv3, acc3);
        a = a_next;
    }

    const int n = lane & 15;
#pragma unroll
    for (int i = 0; i < XG_NT; ++i) {
        const v8f acc = (i == 0) ? acc0 : (i == 1) ? acc1 : (i == 2) ? acc2 : acc3;
        const int wg = wg0 + i;
        const float bias = b_ih[gate * S_ + wg * UNITS + n];
#pragma unroll
        for (int r = 0; r < 8; ++r) {
            const int m = 8 * half + r;
            const size_t t = (size_t)(tbase + m);
            xg[(((t * NWG + wg) * 3 + gate) * B_ + b) * UNITS + n] = (bf16)(acc[r] + bias);
        }
    }
}

// ---- shared pieces for the recurrent kernels -------------------------------

__device__ __forceinline__ void grid_barrier(unsigned* ctr, int t, int tid) {
    __syncthreads();
    __threadfence();
    if (tid == 0) {
        __hip_atomic_fetch_add(ctr, 1u, __ATOMIC_RELEASE, __HIP_MEMORY_SCOPE_AGENT);
        const unsigned target = (unsigned)(t + 1) * NWG;
        while (__hip_atomic_load(ctr, __ATOMIC_ACQUIRE, __HIP_MEMORY_SCOPE_AGENT) < target) {
            __builtin_amdgcn_s_sleep(1);
        }
    }
    __syncthreads();
}

// Copy this WG's weight slice into LDS (b128 chunks).
__device__ __forceinline__ void stage_weights_lds(bf16* __restrict__ wlds,
                                                  const bf16* __restrict__ wpack,
                                                  int first_pg, int npg,
                                                  int wg, int tid) {
    const size_t wtile = (size_t)KTILES * 512;
    for (int pg = 0; pg < npg; ++pg) {
        const v8bf* src = (const v8bf*)(wpack + ((size_t)(first_pg + pg) * NWG + wg) * wtile);
        v8bf* dst = (v8bf*)(wlds + (size_t)pg * KTILES * 512);
        for (int c = tid; c < KTILES * 512 / 8; c += 256) dst[c] = src[c];
    }
    __syncthreads();
}

// ---------------------------------------------------------------------------
// Recurrent kernel, big-ws path: only h-projection in the loop.
// W_hh slice (96 KB) staged in LDS once -> steady-state B-operands come from
// ds_load_b128, keeping L2 free for the h ring + xg stream. 8 waves:
// wave=(khalf<<2)|mt; K split 2-way per M-tile, partials merged in LDS.
// ---------------------------------------------------------------------------
__launch_bounds__(256, 1)
__global__ void gru_recur_xg(const bf16* __restrict__ xg,
                             const float* __restrict__ b_hh,
                             const bf16* __restrict__ wpack,
                             bf16* __restrict__ hbuf,
                             unsigned* __restrict__ ctr) {
    extern __shared__ char smem[];
    bf16*  wlds = (bf16*)smem;                               // 96 KB
    float* glds = (float*)(smem + WSLICE_ELEMS * 2);         // 24 KB [slot][m][n]

    const int tid   = threadIdx.x;
    const int wave  = tid >> 5;
    const int lane  = tid & 31;
    const int khalf = wave >> 2;
    const int mt    = wave & 3;
    const int rowb  = mt * 16 + (lane & 15);
    const int half  = lane >> 4;
    const int wg    = blockIdx.x;

    stage_weights_lds(wlds, wpack, 3, 3, wg, tid);   // W_hh block (pg 3..5)

    for (int t = 0; t < T_; ++t) {
        const bf16* hcur = hbuf + (size_t)(t % 3) * (B_ * S_);
        bf16*       hnxt = hbuf + (size_t)((t + 1) % 3) * (B_ * S_);
        const bf16* hrow = hcur + (size_t)rowb * S_;

        // prefetch next timestep's x-gates block (6 KB) behind this step's math
        if (t + 1 < T_ && tid < 48) {
            const char* xgn = (const char*)(xg + ((size_t)(t + 1) * NWG + wg) * (3 * B_ * UNITS));
            __builtin_prefetch(xgn + tid * 128, 0, 1);
        }

        v8f acc0 = {}, acc1 = {}, acc2 = {};
        const int ktbeg = khalf * (KTILES / 2), ktend = ktbeg + (KTILES / 2);
        v16bf a = load_a_bf16(hrow, ktbeg * 32 + half * 8);
        for (int kt = ktbeg; kt < ktend; ++kt) {
            const bf16* bp = wlds + (size_t)kt * 512 + lane * 16;
            v16bf bv0 = load_b_tile(bp);
            v16bf bv1 = load_b_tile(bp + (size_t)1 * KTILES * 512);
            v16bf bv2 = load_b_tile(bp + (size_t)2 * KTILES * 512);
            v16bf a_next = a;
            if (kt + 1 < ktend) a_next = load_a_bf16(hrow, (kt + 1) * 32 + half * 8);
            acc0 = WMMA_BF16(a, bv0, acc0);
            acc1 = WMMA_BF16(a, bv1, acc1);
            acc2 = WMMA_BF16(a, bv2, acc2);
            a = a_next;
        }
        {
            const int n = lane & 15;
#pragma unroll
            for (int r = 0; r < 8; ++r) {
                const int m = mt * 16 + r + 8 * half;
                glds[((khalf * 3 + 0) * B_ + m) * UNITS + n] = acc0[r];
                glds[((khalf * 3 + 1) * B_ + m) * UNITS + n] = acc1[r];
                glds[((khalf * 3 + 2) * B_ + m) * UNITS + n] = acc2[r];
            }
        }
        __syncthreads();

        const bf16* xgt = xg + ((size_t)t * NWG + wg) * (3 * B_ * UNITS);
#pragma unroll
        for (int e = 0; e < 4; ++e) {
            const int idx = tid * 4 + e;
            const int b   = idx >> 4;
            const int n   = idx & 15;
            const int j   = wg * UNITS + n;
            const float xr = (float)xgt[(0 * B_ + b) * UNITS + n];   // bias pre-folded
            const float xz = (float)xgt[(1 * B_ + b) * UNITS + n];
            const float xn = (float)xgt[(2 * B_ + b) * UNITS + n];
            const float hr = glds[(0 * B_ + b) * UNITS + n] + glds[(3 * B_ + b) * UNITS + n] + b_hh[j];
            const float hz = glds[(1 * B_ + b) * UNITS + n] + glds[(4 * B_ + b) * UNITS + n] + b_hh[S_ + j];
            const float hn = glds[(2 * B_ + b) * UNITS + n] + glds[(5 * B_ + b) * UNITS + n] + b_hh[2 * S_ + j];
            const float r  = sigmoidf_(xr + hr);
            const float z  = sigmoidf_(xz + hz);
            const float nn = tanhf(xn + r * hn);
            const float hp = (float)hcur[(size_t)b * S_ + j];
            hnxt[(size_t)b * S_ + j] = (bf16)((1.0f - z) * nn + z * hp);
        }
        grid_barrier(ctr, t, tid);
    }
}

// ---------------------------------------------------------------------------
// Recurrent kernel, fallback (small ws): fused x+h projections per step.
// Both weight slices (192 KB) staged in LDS; K-loop specialized per proj.
// ---------------------------------------------------------------------------
__launch_bounds__(256, 1)
__global__ void gru_recur_fused(const float* __restrict__ batch,
                                const float* __restrict__ b_ih,
                                const float* __restrict__ b_hh,
                                const bf16* __restrict__ wpack,
                                bf16* __restrict__ hbuf,
                                unsigned* __restrict__ ctr) {
    extern __shared__ char smem[];
    bf16*  wlds = (bf16*)smem;                               // 192 KB (pg 0..5)
    float* glds = (float*)(smem + 2 * WSLICE_ELEMS * 2);     // 24 KB

    const int tid  = threadIdx.x;
    const int wave = tid >> 5;
    const int lane = tid & 31;
    const int proj = wave >> 2;
    const int mt   = wave & 3;
    const int rowb = mt * 16 + (lane & 15);
    const int half = lane >> 4;
    const int wg   = blockIdx.x;

    stage_weights_lds(wlds, wpack, 0, 6, wg, tid);           // W_ih + W_hh

    const bf16* wb = wlds + (size_t)(proj * 3) * KTILES * 512 + lane * 16;

    for (int t = 0; t < T_; ++t) {
        const bf16* hcur = hbuf + (size_t)(t % 3) * (B_ * S_);
        bf16*       hnxt = hbuf + (size_t)((t + 1) % 3) * (B_ * S_);

        v8f acc0 = {}, acc1 = {}, acc2 = {};
        if (proj == 0) {
            const float* xrow = batch + ((size_t)rowb * T_ + t) * S_;
            if (t + 1 < T_) __builtin_prefetch(xrow + S_, 0, 1);   // next timestep
            v16bf a = load_a_f32(xrow, half * 8);
            for (int kt = 0; kt < KTILES; ++kt) {
                const bf16* bp = wb + (size_t)kt * 512;
                v16bf bv0 = load_b_tile(bp);
                v16bf bv1 = load_b_tile(bp + (size_t)1 * KTILES * 512);
                v16bf bv2 = load_b_tile(bp + (size_t)2 * KTILES * 512);
                v16bf a_next = a;
                if (kt + 1 < KTILES) a_next = load_a_f32(xrow, (kt + 1) * 32 + half * 8);
                acc0 = WMMA_BF16(a, bv0, acc0);
                acc1 = WMMA_BF16(a, bv1, acc1);
                acc2 = WMMA_BF16(a, bv2, acc2);
                a = a_next;
            }
        } else {
            const bf16* hrow = hcur + (size_t)rowb * S_;
            v16bf a = load_a_bf16(hrow, half * 8);
            for (int kt = 0; kt < KTILES; ++kt) {
                const bf16* bp = wb + (size_t)kt * 512;
                v16bf bv0 = load_b_tile(bp);
                v16bf bv1 = load_b_tile(bp + (size_t)1 * KTILES * 512);
                v16bf bv2 = load_b_tile(bp + (size_t)2 * KTILES * 512);
                v16bf a_next = a;
                if (kt + 1 < KTILES) a_next = load_a_bf16(hrow, (kt + 1) * 32 + half * 8);
                acc0 = WMMA_BF16(a, bv0, acc0);
                acc1 = WMMA_BF16(a, bv1, acc1);
                acc2 = WMMA_BF16(a, bv2, acc2);
                a = a_next;
            }
        }
        {
            const int n = lane & 15;
#pragma unroll
            for (int r = 0; r < 8; ++r) {
                const int m = mt * 16 + r + 8 * half;
                glds[((proj * 3 + 0) * B_ + m) * UNITS + n] = acc0[r];
                glds[((proj * 3 + 1) * B_ + m) * UNITS + n] = acc1[r];
                glds[((proj * 3 + 2) * B_ + m) * UNITS + n] = acc2[r];
            }
        }
        __syncthreads();

#pragma unroll
        for (int e = 0; e < 4; ++e) {
            const int idx = tid * 4 + e;
            const int b   = idx >> 4;
            const int n   = idx & 15;
            const int j   = wg * UNITS + n;
            const float xr = glds[(0 * B_ + b) * UNITS + n] + b_ih[j];
            const float xz = glds[(1 * B_ + b) * UNITS + n] + b_ih[S_ + j];
            const float xn = glds[(2 * B_ + b) * UNITS + n] + b_ih[2 * S_ + j];
            const float hr = glds[(3 * B_ + b) * UNITS + n] + b_hh[j];
            const float hz = glds[(4 * B_ + b) * UNITS + n] + b_hh[S_ + j];
            const float hn = glds[(5 * B_ + b) * UNITS + n] + b_hh[2 * S_ + j];
            const float r  = sigmoidf_(xr + hr);
            const float z  = sigmoidf_(xz + hz);
            const float nn = tanhf(xn + r * hn);
            const float hp = (float)hcur[(size_t)b * S_ + j];
            hnxt[(size_t)b * S_ + j] = (bf16)((1.0f - z) * nn + z * hp);
        }
        grid_barrier(ctr, t, tid);
    }
}

// Output head: out[b] = sigmoid(h_T[b] . W_out + b_out)
__global__ void out_head(const bf16* __restrict__ hT,
                         const float* __restrict__ Wout,
                         const float* __restrict__ bout,
                         float* __restrict__ out) {
    __shared__ float red[256];
    const int b = blockIdx.x;
    float s = 0.0f;
    for (int j = threadIdx.x; j < S_; j += 256)
        s += (float)hT[(size_t)b * S_ + j] * Wout[j];
    red[threadIdx.x] = s;
    __syncthreads();
    for (int off = 128; off > 0; off >>= 1) {
        if (threadIdx.x < off) red[threadIdx.x] += red[threadIdx.x + off];
        __syncthreads();
    }
    if (threadIdx.x == 0) out[b] = sigmoidf_(red[0] + bout[0]);
}

extern "C" void kernel_launch(void* const* d_in, const int* in_sizes, int n_in,
                              void* d_out, int out_size, void* d_ws, size_t ws_size,
                              hipStream_t stream) {
    const float* batch = (const float*)d_in[0];
    const float* Wih   = (const float*)d_in[1];
    const float* Whh   = (const float*)d_in[2];
    const float* bih   = (const float*)d_in[3];
    const float* bhh   = (const float*)d_in[4];
    const float* Wout  = (const float*)d_in[5];
    const float* bout  = (const float*)d_in[6];
    float* out = (float*)d_out;

    char* ws = (char*)d_ws;
    unsigned* ctr = (unsigned*)(ws + WS_CTR_OFF);
    bf16* hbuf    = (bf16*)(ws + WS_H_OFF);
    bf16* wpack   = (bf16*)(ws + WS_W_OFF);
    bf16* xg      = (bf16*)(ws + WS_XG_OFF);

    init_state<<<(3 * B_ * S_ + 255) / 256, 256, 0, stream>>>(hbuf, ctr);
    pack_weights<<<(unsigned)((WPACK_ELEMS + 255) / 256), 256, 0, stream>>>(Wih, Whh, wpack);

    if (ws_size >= (size_t)WS_NEED_BIG) {
        // Phase 1: all-timestep input projection at full-chip WMMA parallelism
        const unsigned tasks = (B_ * T_ / 16) * (3 * S_ / 16 / XG_NT);   // 196608
        xg_gemm<<<tasks / 8, 256, 0, stream>>>(batch, bih, wpack, xg);
        // Phase 2: latency-lean recurrence; 96 KB W_hh slice + 24 KB gates in LDS
        const size_t lds2 = (size_t)WSLICE_ELEMS * 2 + GATE_F32 * 4;     // 120 KB
        gru_recur_xg<<<NWG, 256, lds2, stream>>>(xg, bhh, wpack, hbuf, ctr);
    } else {
        // Fallback: fused x+h recurrence; 192 KB weights + 24 KB gates in LDS
        const size_t ldsf = (size_t)2 * WSLICE_ELEMS * 2 + GATE_F32 * 4; // 216 KB
        gru_recur_fused<<<NWG, 256, ldsf, stream>>>(batch, bih, bhh, wpack, hbuf, ctr);
    }

    const bf16* hT = hbuf + (size_t)(T_ % 3) * (B_ * S_);
    out_head<<<B_, 256, 0, stream>>>(hT, Wout, bout, out);
}
